// ScaledDotProductAttention_23235773071932
// MI455X (gfx1250) — matrix operand order repaired and checked by reading, not compiled, not run on hardware
//
#include <hip/hip_runtime.h>
#include <hip/hip_bf16.h>

// ---------------------------------------------------------------------------
// Scaled dot-product attention for MI455X (gfx1250, wave32).
// Compute-bound (~47 GFLOP vs ~50 MB traffic @ 23.3 TB/s) => all matmuls on
// v_wmma_f32_16x16x32_bf16 with f32 accumulation.
// Stage 1: QKV projection (f32 -> bf16 in workspace); V is stored TRANSPOSED
//          (Vt[e][row]) so the attention P*V B-fragments are contiguous.
// Stage 2: flash attention; K/V tiles staged by the Tensor Data Mover.
//          TDM ops from one wave retire in order, so s_wait_tensorcnt(1)
//          releases the K tile while the V DMA overlaps the score matmuls;
//          s_wait_tensorcnt(0) fences V right before the P*V update.
// ---------------------------------------------------------------------------

typedef __attribute__((ext_vector_type(16))) __bf16 v16bf;
typedef __attribute__((ext_vector_type(8)))  float  v8f;
typedef unsigned int v4u __attribute__((ext_vector_type(4)));
typedef int          v4i __attribute__((ext_vector_type(4)));
typedef int          v8i __attribute__((ext_vector_type(8)));

#define DMODEL 512
#define NROWS  8192          // B*N = 4*2048 flattened rows
#define SEQ    2048
#define KVTILE 128

#if defined(__has_builtin)
#if __has_builtin(__builtin_amdgcn_tensor_load_to_lds) && \
    __has_builtin(__builtin_amdgcn_s_wait_tensorcnt)
#define USE_TDM 1
#endif
#endif
#ifndef USE_TDM
#define USE_TDM 0
#endif

// ---- WMMA fragment loaders (per ISA 7.12.2 wave32 layouts) ----------------

// A-matrix 16x32 bf16. Lane L: M = L%16. Element j: v=j/2,
// K = (v/4)*16 + (L/16)*8 + (v&3)*2 + (j&1).  -> paired ds_load_b128.
__device__ __forceinline__ v16bf frag_a_bf16(const __bf16* base, int stride) {
  const int lane = threadIdx.x & 31;
  const int m = lane & 15, hi = lane >> 4;
  v16bf f;
#pragma unroll
  for (int j = 0; j < 16; ++j) {
    const int v = j >> 1;
    const int k = ((v >> 2) << 4) + (hi << 3) + ((v & 3) << 1) + (j & 1);
    f[j] = base[m * stride + k];
  }
  return f;
}

// B-matrix 32x16 bf16, source row-major with rows == B columns
// (B[k][n] = base[n*stride + k]).  Lane L: N = L%16, elem j: K = (j/8)*16 + (L/16)*8 + (j%8).
// two runs of 8 consecutive bf16 per lane -> paired ds_load_b128.
__device__ __forceinline__ v16bf frag_b_rows(const __bf16* base, int stride) {
  const int lane = threadIdx.x & 31;
  const int n = lane & 15, hi = lane >> 4;
  v16bf f;
#pragma unroll
  for (int j = 0; j < 16; ++j) {
    const int k = ((j >> 3) << 4) + (hi << 3) + (j & 7);
    f[j] = base[n * stride + k];
  }
  return f;
}

#if USE_TDM
// ---------------------------------------------------------------------------
// Tensor Data Mover: 2D bf16 tile load, global -> LDS (ISA ch.8 D# layout).
// group0: count=1 | lds_addr | global_addr[56:0] | type=2
// group1: data_size=1 (2B), tensor_dim0=cols, tensor_dim1=rows,
//         tile_dim0=cols, tile_dim1=rows, tensor_dim0_stride=row_stride.
// Issue from ONE wave (EXEC is ignored by TDM); fence with s_wait_tensorcnt.
// ---------------------------------------------------------------------------
__device__ __forceinline__ void tdm_load_2d(unsigned lds_off, const void* gptr,
                                            unsigned rows, unsigned cols,
                                            unsigned row_stride) {
  const unsigned long long ga = (unsigned long long)gptr;
  v4u g0;
  g0.x = 1u;                                                 // count=1
  g0.y = lds_off;                                            // lds_addr
  g0.z = (unsigned)ga;                                       // global_addr lo
  g0.w = (unsigned)((ga >> 32) & 0x01ffffffu) | (2u << 30);  // ga hi | type=2
  v8i g1;
  g1[0] = (int)(1u << 16);                     // data_size=2B, wg_mask=0
  g1[1] = (int)(cols << 16);                   // tensor_dim0[15:0]
  g1[2] = (int)((rows << 16) | (cols >> 16));  // tensor_dim1 lo | dim0 hi
  g1[3] = (int)((cols << 16) | (rows >> 16));  // tile_dim0 | tensor_dim1 hi
  g1[4] = (int)(rows & 0xffffu);               // tile_dim1 (tile_dim2 = 0)
  g1[5] = (int)row_stride;                     // tensor_dim0_stride lo
  g1[6] = 0;                                   // stride0 hi | stride1 lo
  g1[7] = 0;                                   // stride1 hi
  const v4i gz = {0, 0, 0, 0};
#if __clang_major__ >= 23
  const v8i gz8 = {0, 0, 0, 0, 0, 0, 0, 0};
  __builtin_amdgcn_tensor_load_to_lds(g0, g1, gz, gz, gz8, 0);
#else
  __builtin_amdgcn_tensor_load_to_lds(g0, g1, gz, gz, 0);
#endif
}
#endif  // USE_TDM

// ---------------------------------------------------------------------------
// Kernel 1: y = x @ W^T + b, emitted as bf16.  blockIdx.z selects Q/K/V.
// Block = 128 threads (4 waves).  Tile: 64 rows x 64 cols, K-step 32.
// V (z==2) is written transposed: Vt[col * NROWS + row].
// ---------------------------------------------------------------------------
__global__ void __launch_bounds__(128)
qkv_proj_kernel(const float* __restrict__ x,
                const float* __restrict__ Wq, const float* __restrict__ bq,
                const float* __restrict__ Wk, const float* __restrict__ bk,
                const float* __restrict__ Wv, const float* __restrict__ bv,
                __bf16* __restrict__ Q, __bf16* __restrict__ K,
                __bf16* __restrict__ Vt) {
  const int z = blockIdx.z;
  const float* W    = (z == 0) ? Wq : ((z == 1) ? Wk : Wv);
  const float* bias = (z == 0) ? bq : ((z == 1) ? bk : bv);
  __bf16* dst       = (z == 0) ? Q  : ((z == 1) ? K  : Vt);
  const bool transposed = (z == 2);

  const int r0 = blockIdx.x * 64;   // output rows (flattened b*n)
  const int e0 = blockIdx.y * 64;   // output cols
  const int tid  = threadIdx.x;
  const int wv   = tid >> 5;
  const int lane = tid & 31;

  __shared__ __bf16 Xs[64 * 32];
  __shared__ __bf16 Ws[64 * 32];

  v8f acc[4] = {};

  for (int dk = 0; dk < DMODEL; dk += 32) {
    for (int i = tid; i < 64 * 32; i += 128) {
      const int row = i >> 5, col = i & 31;
      Xs[i] = (__bf16)x[(size_t)(r0 + row) * DMODEL + dk + col];
      Ws[i] = (__bf16)W[(size_t)(e0 + row) * DMODEL + dk + col];
    }
    __syncthreads();

    const v16bf a = frag_a_bf16(&Xs[(16 * wv) * 32], 32);
#pragma unroll
    for (int t = 0; t < 4; ++t) {
      const v16bf b = frag_b_rows(&Ws[(16 * t) * 32], 32);
      acc[t] = __builtin_amdgcn_wmma_f32_16x16x32_bf16(
          false, a, false, b, (short)0, acc[t], false, false);
    }
    __syncthreads();
  }

  const int hi = lane >> 4, n = lane & 15;
#pragma unroll
  for (int t = 0; t < 4; ++t) {
#pragma unroll
    for (int r = 0; r < 8; ++r) {
      const int row = r0 + 16 * wv + r + 8 * hi;
      const int col = e0 + 16 * t + n;
      const __bf16 v = (__bf16)(acc[t][r] + bias[col]);
      if (transposed)
        dst[(size_t)col * NROWS + row] = v;    // Vt[e][row]
      else
        dst[(size_t)row * DMODEL + col] = v;   // Q/K[row][e]
    }
  }
}

// ---------------------------------------------------------------------------
// Kernel 2: flash attention.  Block = 256 threads (8 waves), 16 query rows.
// LDS: Q 16x512, K 128x512 (row-major), Vt 512x128 (e-major), S/P/stats.
// Wave w: score tile (16q x keys[16w..16w+16)) with full D=512 reduction,
// then owns output columns [64w, 64w+64) for the P*V update.
// ---------------------------------------------------------------------------
// Dynamic-LDS byte offsets (no static __shared__ here, so the dynamic
// segment starts at LDS offset 0 -> these double as D# lds_addr values).
#define SM_QS    0
#define SM_KS    16384
#define SM_VS    147456
#define SM_S     278528
#define SM_P     286720
#define SM_M     290816
#define SM_L     290880
#define SM_ALPHA 290944
#define SMEM_BYTES 291008

__global__ void __launch_bounds__(256)
flash_attn_kernel(const __bf16* __restrict__ Q, const __bf16* __restrict__ K,
                  const __bf16* __restrict__ Vt, float* __restrict__ out) {
  extern __shared__ char smem[];
  __bf16* Qs    = (__bf16*)(smem + SM_QS);   // 16 x 512
  __bf16* Ks    = (__bf16*)(smem + SM_KS);   // 128 x 512 (key-major)
  __bf16* Vs    = (__bf16*)(smem + SM_VS);   // 512 x 128 (e-major)
  float*  S     = (float*)(smem + SM_S);     // 16 x 128
  __bf16* P     = (__bf16*)(smem + SM_P);    // 16 x 128
  float*  row_m = (float*)(smem + SM_M);
  float*  row_l = (float*)(smem + SM_L);
  float*  row_a = (float*)(smem + SM_ALPHA);

  const int tid  = threadIdx.x;
  const int wv   = tid >> 5;
  const int lane = tid & 31;
  const int hi   = lane >> 4, n = lane & 15;
  const int r0   = blockIdx.x * 16;       // global query row
  const int batch = r0 >> 11;             // r0 / 2048
  const int kvrow0 = batch * SEQ;         // first key row of this batch
  const float scale = 0.044194173824159216f;  // 1/sqrt(512)

  // ---- Stage Q tile (16 x 512 bf16) ----
#if USE_TDM
  if (wv == 0) {   // scalar branch: TDM ignores EXEC, so issue from one wave
    tdm_load_2d(SM_QS, Q + (size_t)r0 * DMODEL, 16, DMODEL, DMODEL);
    __builtin_amdgcn_s_wait_tensorcnt(0);
  }
#else
  {
    const uint4* src = (const uint4*)(Q + (size_t)r0 * DMODEL);
    uint4* dq = (uint4*)Qs;
    for (int i = tid; i < 16 * DMODEL / 8; i += 256) dq[i] = src[i];
  }
#endif
  if (tid < 16) { row_m[tid] = -3.0e38f; row_l[tid] = 0.0f; }

  v8f acc[4] = {};
  __syncthreads();

  for (int kt = 0; kt < SEQ / KVTILE; ++kt) {
    const int krow = kvrow0 + kt * KVTILE;           // first key of tile

    // ---- Issue K then V DMA.  In-order retire => tensorcnt<=1 means the K
    // tile has landed; the V (transposed, strided) DMA keeps running while
    // the score matmuls execute. ----
#if USE_TDM
    if (wv == 0) {
      tdm_load_2d(SM_KS, K + (size_t)krow * DMODEL, KVTILE, DMODEL, DMODEL);
      tdm_load_2d(SM_VS, Vt + krow, DMODEL, KVTILE, NROWS);  // 512 rows x 128
      __builtin_amdgcn_s_wait_tensorcnt(1);   // K ready, V in flight
    }
#else
    {
      const uint4* ksrc = (const uint4*)(K + (size_t)krow * DMODEL);
      uint4* kd = (uint4*)Ks;
      for (int i = tid; i < KVTILE * DMODEL / 8; i += 256) kd[i] = ksrc[i];
      const uint4* vsrc = (const uint4*)(Vt + krow);
      uint4* vd = (uint4*)Vs;
      for (int i = tid; i < DMODEL * (KVTILE / 8); i += 256) {
        const int e = i / (KVTILE / 8), c = i % (KVTILE / 8);
        vd[e * (KVTILE / 8) + c] = vsrc[(size_t)e * (NROWS / 8) + c];
      }
    }
#endif
    if (kt + 1 < SEQ / KVTILE) {   // global_prefetch_b8 the next tiles (->L2)
      __builtin_prefetch(K + (size_t)(krow + KVTILE) * DMODEL, 0, 0);
      __builtin_prefetch(Vt + krow + KVTILE, 0, 0);
    }
    __syncthreads();

    // ---- Scores: 16 x 16 tile per wave, full D=512 reduction ----
    {
      v8f sacc = {0.f, 0.f, 0.f, 0.f, 0.f, 0.f, 0.f, 0.f};
#pragma unroll 4
      for (int dk = 0; dk < DMODEL; dk += 32) {
        const v16bf a = frag_a_bf16(Qs + dk, DMODEL);
        const v16bf b = frag_b_rows(Ks + (size_t)(16 * wv) * DMODEL + dk, DMODEL);
        sacc = __builtin_amdgcn_wmma_f32_16x16x32_bf16(
            false, a, false, b, (short)0, sacc, false, false);
      }
#pragma unroll
      for (int r = 0; r < 8; ++r)
        S[(r + 8 * hi) * KVTILE + 16 * wv + n] = sacc[r] * scale;
    }
    __syncthreads();

    // ---- Online softmax, one lane per query row ----
    if (tid < 16) {
      const float* srow = S + tid * KVTILE;
      float mprev = row_m[tid];
      float tmax = -3.0e38f;
      for (int j = 0; j < KVTILE; ++j) tmax = fmaxf(tmax, srow[j]);
      const float mnew  = fmaxf(mprev, tmax);
      const float alpha = __expf(mprev - mnew);
      float lsum = 0.0f;
      __bf16* prow = P + tid * KVTILE;
      for (int j = 0; j < KVTILE; ++j) {
        const float p = __expf(srow[j] - mnew);
        prow[j] = (__bf16)p;
        lsum += p;
      }
      row_l[tid] = alpha * row_l[tid] + lsum;
      row_m[tid] = mnew;
      row_a[tid] = alpha;
    }
#if USE_TDM
    if (wv == 0) __builtin_amdgcn_s_wait_tensorcnt(0);  // V tile landed
#endif
    __syncthreads();

    // ---- Rescale accumulators by alpha(row), then O += P x V ----
#pragma unroll
    for (int t = 0; t < 4; ++t)
#pragma unroll
      for (int r = 0; r < 8; ++r)
        acc[t][r] *= row_a[r + 8 * hi];

    const int c0 = 64 * wv;
#pragma unroll
    for (int kk = 0; kk < 4; ++kk) {
      const v16bf a = frag_a_bf16(P + kk * 32, KVTILE);
#pragma unroll
      for (int t = 0; t < 4; ++t) {
        // B[k][n] = V[key][e] = Vs[(c0+16t+n)*128 + kk*32 + k]  (contiguous k)
        const v16bf b = frag_b_rows(Vs + (size_t)(c0 + 16 * t) * KVTILE + kk * 32,
                                    KVTILE);
        acc[t] = __builtin_amdgcn_wmma_f32_16x16x32_bf16(
            false, a, false, b, (short)0, acc[t], false, false);
      }
    }
    __syncthreads();   // before next tile overwrites Ks/Vs
  }

  // ---- Epilogue: divide by softmax denominator, store f32 ----
#pragma unroll
  for (int t = 0; t < 4; ++t) {
#pragma unroll
    for (int r = 0; r < 8; ++r) {
      const int row = r + 8 * hi;
      const float o = acc[t][r] / row_l[row];
      out[(size_t)(r0 + row) * DMODEL + 64 * wv + 16 * t + n] = o;
    }
  }
}

// ---------------------------------------------------------------------------
extern "C" void kernel_launch(void* const* d_in, const int* in_sizes, int n_in,
                              void* d_out, int out_size, void* d_ws,
                              size_t ws_size, hipStream_t stream) {
  const float* x  = (const float*)d_in[0];
  const float* Wq = (const float*)d_in[1];
  const float* bq = (const float*)d_in[2];
  const float* Wk = (const float*)d_in[3];
  const float* bk = (const float*)d_in[4];
  const float* Wv = (const float*)d_in[5];
  const float* bv = (const float*)d_in[6];

  __bf16* Qw  = (__bf16*)d_ws;
  __bf16* Kw  = Qw + (size_t)NROWS * DMODEL;
  __bf16* Vwt = Kw + (size_t)NROWS * DMODEL;   // transposed: Vt[e][row]

  dim3 gproj(NROWS / 64, DMODEL / 64, 3);
  qkv_proj_kernel<<<gproj, 128, 0, stream>>>(x, Wq, bq, Wk, bk, Wv, bv,
                                             Qw, Kw, Vwt);

  dim3 gattn(NROWS / 16);
  flash_attn_kernel<<<gattn, 256, SMEM_BYTES, stream>>>(Qw, Kw, Vwt,
                                                        (float*)d_out);
}
